// HOANLayer_89953795048031
// MI455X (gfx1250) — matrix operand order repaired and checked by reading, not compiled, not run on hardware
//
#include <hip/hip_runtime.h>
#include <hip/hip_bf16.h>
#include <cstdint>

typedef __attribute__((ext_vector_type(16))) _Float16 v16h;
typedef __attribute__((ext_vector_type(8)))  float    v8f;

#define NROW 8192
#define DIN  128
#define DOUT 64
#define NTILES (NROW / 32)
#define DEPTH 4   // async pipeline depth (tiles in flight)

// ---------------------------------------------------------------------------
// Kernel 1: xw = x @ w  (8192x128 @ 128x64), write f32 + f16 copies.
// ---------------------------------------------------------------------------
__global__ __launch_bounds__(256) void k_xw(const float* __restrict__ x,
                                            const float* __restrict__ w,
                                            float* __restrict__ outf,
                                            _Float16* __restrict__ outh) {
    int idx = blockIdx.x * 256 + threadIdx.x;   // NROW*DOUT threads
    int i = idx >> 6;
    int k = idx & 63;
    const float* xr = x + (size_t)i * DIN;
    float acc = 0.f;
#pragma unroll 8
    for (int c = 0; c < DIN; ++c) acc = fmaf(xr[c], w[c * DOUT + k], acc);
    outf[idx] = acc;
    outh[idx] = (_Float16)acc;
}

// ---------------------------------------------------------------------------
// Kernel 2: s1 = xw_s@a1, s2 = xw_s@a2, t1 = xw_t@a1, t2 = xw_t@a2
// ---------------------------------------------------------------------------
__global__ __launch_bounds__(256) void k_dots(const float* __restrict__ xw_s,
                                              const float* __restrict__ xw_t,
                                              const float* __restrict__ a,
                                              float* __restrict__ s1, float* __restrict__ s2,
                                              float* __restrict__ t1, float* __restrict__ t2) {
    int i = blockIdx.x * 256 + threadIdx.x;
    const float* rs = xw_s + (size_t)i * DOUT;
    const float* rt = xw_t + (size_t)i * DOUT;
    float as1 = 0.f, as2 = 0.f, at1 = 0.f, at2 = 0.f;
#pragma unroll 8
    for (int c = 0; c < DOUT; ++c) {
        float a1 = a[c], a2 = a[DOUT + c];
        as1 = fmaf(rs[c], a1, as1);
        as2 = fmaf(rs[c], a2, as2);
        at1 = fmaf(rt[c], a1, at1);
        at2 = fmaf(rt[c], a2, at2);
    }
    s1[i] = as1; s2[i] = as2; t1[i] = at1; t2[i] = at2;
}

// ---------------------------------------------------------------------------
// Kernel 3: global maxima of t2 and s2 (safe softmax shift bounds)
// ---------------------------------------------------------------------------
__global__ __launch_bounds__(256) void k_max(const float* __restrict__ t2,
                                             const float* __restrict__ s2,
                                             float* __restrict__ maxes) {
    __shared__ float sm0[256], sm1[256];
    int tid = threadIdx.x;
    float m0 = -1e30f, m1 = -1e30f;
    for (int i = tid; i < NROW; i += 256) {
        m0 = fmaxf(m0, t2[i]);
        m1 = fmaxf(m1, s2[i]);
    }
    sm0[tid] = m0; sm1[tid] = m1;
    __syncthreads();
    for (int s = 128; s > 0; s >>= 1) {
        if (tid < s) {
            sm0[tid] = fmaxf(sm0[tid], sm0[tid + s]);
            sm1[tid] = fmaxf(sm1[tid], sm1[tid + s]);
        }
        __syncthreads();
    }
    if (tid == 0) { maxes[0] = sm0[0]; maxes[1] = sm1[0]; }
}

// issue one 16-byte async memory->LDS copy per lane (tracked by ASYNCcnt)
#define ASYNC_LOAD_B128(ldsoff, gaddr)                                    \
    asm volatile("global_load_async_to_lds_b128 %0, %1, off"              \
                 :: "v"(ldsoff), "v"(gaddr) : "memory")

// ---------------------------------------------------------------------------
// Kernel 4: fused masked-softmax attention aggregation (flash-style, WMMA,
// async-to-LDS quad-buffered adjacency streaming).
//   out[i,:] = elu( softmax_j( mask(i,j) ? leaky(svec[i]+tvec[j]) : -inf )
//                   @ xw[j,:] + bias )
// TRANS=false: mask(i,j) = adj[i][j]   (update_source)   tile in LDS as [m][k]
// TRANS=true : mask(i,j) = adj[j][i]   (update_target)   tile in LDS as [k][m]
// One wave owns a 16-row strip; j streamed in tiles of 32; waves independent.
// ---------------------------------------------------------------------------
template <bool TRANS>
__global__ __launch_bounds__(256) void k_attn(const int* __restrict__ adj,
                                              const float* __restrict__ svec,
                                              const float* __restrict__ tvec,
                                              const float* __restrict__ maxp,
                                              const _Float16* __restrict__ xwh,
                                              const float* __restrict__ bias,
                                              float* __restrict__ out) {
    __shared__ __attribute__((aligned(16))) int s_adj[8][DEPTH][512]; // 2KB tiles
    __shared__ float s_rs[8][16];

    const int lane = threadIdx.x & 31;
    const int wave = threadIdx.x >> 5;
    const int strip = blockIdx.x * 8 + wave;
    const int i0 = strip * 16;
    const int m = lane & 15;     // A row within strip; B/C column for this lane
    const int H = lane >> 4;     // lane half

    const float sv = svec[i0 + m];
    const float gmax = maxp[0];
    const float z0 = sv + gmax;
    const float mi = z0 > 0.f ? z0 : 0.01f * z0;   // per-row softmax shift (upper bound)

    // --- per-lane async staging addresses: 4 x 16B segments per tile ---
    const unsigned ldsWave =
        (unsigned)(uintptr_t)(&s_adj[wave][0][0]) + (unsigned)lane * 16u;
    unsigned long long ga[4];
    const unsigned long long delta =
        TRANS ? (unsigned long long)NROW * 4ull * 32ull : 128ull;
#pragma unroll
    for (int q = 0; q < 4; ++q) {
        int sidx = q * 32 + lane;
        if (TRANS) {
            int r = sidx >> 2, seg = sidx & 3;        // r: j-local row, 16-int rows
            ga[q] = (unsigned long long)(uintptr_t)adj +
                    ((unsigned long long)r * NROW + (unsigned)(i0 + seg * 4)) * 4ull;
        } else {
            int r = sidx >> 3, seg = sidx & 7;        // r: i-local row, 32-int rows
            ga[q] = (unsigned long long)(uintptr_t)adj +
                    ((unsigned long long)(i0 + r) * NROW + (unsigned)(seg * 4)) * 4ull;
        }
    }

    // --- prologue: fill DEPTH-1 tiles of the pipeline ---
#pragma unroll
    for (int p = 0; p < DEPTH - 1; ++p) {
#pragma unroll
        for (int q = 0; q < 4; ++q)
            ASYNC_LOAD_B128(ldsWave + (unsigned)p * 2048u + (unsigned)q * 512u, ga[q]);
#pragma unroll
        for (int q = 0; q < 4; ++q) ga[q] += delta;
    }

    v8f acc[4] = {};             // 4 x (16x16 f32) C tiles -> 64 output features
    float ps0 = 0.f, ps1 = 0.f, ps2 = 0.f, ps3 = 0.f;  // parallel row-sum partials

    for (int it = 0; it < NTILES; ++it) {
        const int buf  = it & (DEPTH - 1);
        const int wbuf = (it + DEPTH - 1) & (DEPTH - 1);

        // buffer wbuf was consumed in iteration it-1: make sure its DS reads
        // retired before the async engine overwrites it.
        asm volatile("s_wait_dscnt 0x0" ::: "memory");
#pragma unroll
        for (int q = 0; q < 4; ++q)
            ASYNC_LOAD_B128(ldsWave + (unsigned)wbuf * 2048u + (unsigned)q * 512u, ga[q]);
        // advance prefetch addresses (clamp at last tile; uniform select, no branch)
        const unsigned long long step = (it + DEPTH <= NTILES - 1) ? delta : 0ull;
#pragma unroll
        for (int q = 0; q < 4; ++q) ga[q] += step;

        // async loads retire in order: <=12 outstanding => tile `it` resident.
        asm volatile("s_wait_asynccnt 0xc" ::: "memory");

        const int j0 = it * 32;

        // ---- issue B loads FIRST: their L2 latency hides behind the A-build ----
        // lane (n=m,H), element e -> K = e + 8*H + (e>=8 ? 8 : 0); four 16-wide feature slices.
        v16h bv0, bv1, bv2, bv3;
#pragma unroll
        for (int e = 0; e < 16; ++e) {
            int K = e + (H << 3) + ((e >= 8) ? 8 : 0);
            const _Float16* br = xwh + (size_t)(j0 + K) * DOUT + m;
            bv0[e] = br[0];
            bv1[e] = br[16];
            bv2[e] = br[32];
            bv3[e] = br[48];
        }

        const float tv = tvec[j0 + lane];   // lane holds t2[j0+lane]

        // ---- build P tile directly in 16x32 f16 A-operand layout ----
        // lane (m,H), element e -> K = e + 8*H + (e>=8 ? 8 : 0)
        v16h av;
#pragma unroll
        for (int e = 0; e < 16; ++e) {
            int K = e + (H << 3) + ((e >= 8) ? 8 : 0);
            int ad = TRANS ? s_adj[wave][buf][K * 16 + m]
                           : s_adj[wave][buf][m * 32 + K];
            float t2K = __shfl(tv, K, 32);               // broadcast, no memory
            float z = sv + t2K;
            float l = z > 0.f ? z : 0.01f * z;           // leaky_relu
            float p = (ad != 0) ? __expf(l - mi) : 0.f;  // masked exp
            if ((e & 3) == 0) ps0 += p;
            else if ((e & 3) == 1) ps1 += p;
            else if ((e & 3) == 2) ps2 += p;
            else ps3 += p;
            av[e] = (_Float16)p;
        }

        // ---- 4 back-to-back WMMAs ----
        acc[0] = __builtin_amdgcn_wmma_f32_16x16x32_f16(
            false, av, false, bv0, (short)0, acc[0], false, false);
        acc[1] = __builtin_amdgcn_wmma_f32_16x16x32_f16(
            false, av, false, bv1, (short)0, acc[1], false, false);
        acc[2] = __builtin_amdgcn_wmma_f32_16x16x32_f16(
            false, av, false, bv2, (short)0, acc[2], false, false);
        acc[3] = __builtin_amdgcn_wmma_f32_16x16x32_f16(
            false, av, false, bv3, (short)0, acc[3], false, false);
    }

    // ---- row-sum reduction: lanes m and m^16 hold the two halves of row m ----
    float psum = (ps0 + ps1) + (ps2 + ps3);
    float full = psum + __shfl_xor(psum, 16, 32);
    if (lane < 16) s_rs[wave][lane] = full;
    __syncthreads();

    // ---- epilogue: normalize, bias, ELU, store (C layout: VGPR c -> M=c+8H) ----
#pragma unroll
    for (int c = 0; c < 8; ++c) {
        int row = c + (H << 3);
        float rinv = 1.0f / s_rs[wave][row];
#pragma unroll
        for (int t = 0; t < 4; ++t) {
            float vv = acc[t][c] * rinv + bias[t * 16 + m];
            float o = vv > 0.f ? vv : expm1f(vv);
            out[(size_t)(i0 + row) * DOUT + t * 16 + m] = o;
        }
    }
}

// ---------------------------------------------------------------------------
extern "C" void kernel_launch(void* const* d_in, const int* in_sizes, int n_in,
                              void* d_out, int out_size, void* d_ws, size_t ws_size,
                              hipStream_t stream) {
    (void)in_sizes; (void)n_in; (void)out_size; (void)ws_size;

    const float* x_s = (const float*)d_in[0];
    const float* x_t = (const float*)d_in[1];
    const int*   adj = (const int*)d_in[2];
    const float* w_s = (const float*)d_in[3];
    const float* w_t = (const float*)d_in[4];
    const float* a   = (const float*)d_in[5];
    const float* b_s = (const float*)d_in[6];
    const float* b_t = (const float*)d_in[7];

    char* ws = (char*)d_ws;
    float*    xw_s   = (float*)ws;    ws += (size_t)NROW * DOUT * 4;
    float*    xw_t   = (float*)ws;    ws += (size_t)NROW * DOUT * 4;
    _Float16* xw_s_h = (_Float16*)ws; ws += (size_t)NROW * DOUT * 2;
    _Float16* xw_t_h = (_Float16*)ws; ws += (size_t)NROW * DOUT * 2;
    float* s1 = (float*)ws; ws += NROW * 4;
    float* s2 = (float*)ws; ws += NROW * 4;
    float* t1 = (float*)ws; ws += NROW * 4;
    float* t2 = (float*)ws; ws += NROW * 4;
    float* maxes = (float*)ws;

    dim3 blk(256);
    k_xw<<<NROW * DOUT / 256, blk, 0, stream>>>(x_s, w_s, xw_s, xw_s_h);
    k_xw<<<NROW * DOUT / 256, blk, 0, stream>>>(x_t, w_t, xw_t, xw_t_h);
    k_dots<<<NROW / 256, blk, 0, stream>>>(xw_s, xw_t, a, s1, s2, t1, t2);
    k_max<<<1, blk, 0, stream>>>(t2, s2, maxes);

    float* outS = (float*)d_out;
    float* outT = outS + (size_t)NROW * DOUT;
    // update_source: mask = adj, logits from (s1, t2), aggregate xw_s
    k_attn<false><<<NROW / 16 / 8, blk, 0, stream>>>(adj, s1, t2, maxes + 0, xw_s_h, b_s, outS);
    // update_target: mask = adj.T, logits from (t1, s2), aggregate xw_t
    k_attn<true><<<NROW / 16 / 8, blk, 0, stream>>>(adj, t1, s2, maxes + 1, xw_t_h, b_t, outT);
}
